// MaskedMultiHeadAttention_67499706024334
// MI455X (gfx1250) — compile-verified, matrix operand order checked by reading, not run on hardware
//
#include <hip/hip_runtime.h>
#include <math.h>
#include <stdint.h>

typedef __bf16 bf16;
typedef __attribute__((ext_vector_type(16))) __bf16 v16bf;
typedef __attribute__((ext_vector_type(8)))  __bf16 v8bf;
typedef __attribute__((ext_vector_type(8)))  float  v8f;

#define Bn  4
#define Sn  2048
#define Dn  1024
#define Hn  16
#define DHn 64
#define Mn  (Bn * Sn)          /* 8192 rows of x */

/* workspace layout (bytes) */
#define XH_OFF  0ull
#define XH_BY   ((size_t)Mn * Dn * 2)        /* x in bf16            16 MiB */
#define WT_OFF  (XH_OFF + XH_BY)
#define WT_BY   ((size_t)3 * Dn * Dn * 2)    /* Wq/Wk/Wv^T bf16       6 MiB */
#define QH_OFF  (WT_OFF + WT_BY)
#define QH_BY   ((size_t)Mn * Dn * 2)        /* Q bf16 (B,H,S,dh)    16 MiB */
#define KH_OFF  (QH_OFF + QH_BY)
#define KH_BY   ((size_t)Mn * Dn * 2)        /* K bf16 (B,H,S,dh)    16 MiB */
#define VT_OFF  (KH_OFF + KH_BY)
#define VT_BY   ((size_t)Mn * Dn * 2)        /* V^T bf16 (B,H,dh,S)  16 MiB */
#define AT_OFF  (VT_OFF + VT_BY)
#define AT_BY   ((size_t)Mn * Dn * 4)        /* attn f32 (B,S,D)     32 MiB */

/* async 16B global -> LDS copy, tracked by ASYNCcnt (CDNA5 VFLAT async).
 * VDST = VGPR holding LDS byte address, VADDR = 64-bit global address. */
__device__ __forceinline__ void async_b128(const bf16* g, bf16* l) {
  asm volatile("global_load_async_to_lds_b128 %0, %1, off"
               :
               : "v"((unsigned)(uintptr_t)l),
                 "v"((unsigned long long)(uintptr_t)g)
               : "memory");
}

__device__ __forceinline__ void wait_async() {
  asm volatile("s_wait_asynccnt 0x0" ::: "memory");
}

__device__ __forceinline__ v16bf load_frag(const bf16* p) {
  union { v16bf v; v8bf h[2]; } f;
  f.h[0] = *(const v8bf*)(p);
  f.h[1] = *(const v8bf*)(p + 16);
  return f.v;
}

__device__ __forceinline__ v8f wmma_bf16(v16bf a, v16bf b, v8f c) {
  return __builtin_amdgcn_wmma_f32_16x16x32_bf16(
      false, a, false, b, (short)0, c, false, false);
}

/* ---------------- conversion kernels ---------------- */

__global__ void cvt_x_kernel(const float* __restrict__ x, bf16* __restrict__ xh) {
  int i = blockIdx.x * 256 + threadIdx.x;
  xh[i] = (bf16)x[i];
}

__global__ void cvt_wT_kernel(const float* __restrict__ Wq,
                              const float* __restrict__ Wk,
                              const float* __restrict__ Wv,
                              bf16* __restrict__ wt) {
  int mat = blockIdx.y;
  const float* W = (mat == 0) ? Wq : ((mat == 1) ? Wk : Wv);
  int i = blockIdx.x * 256 + threadIdx.x;   /* i = n*1024 + k */
  int n = i >> 10, k = i & 1023;
  wt[(size_t)mat * Dn * Dn + i] = (bf16)W[k * Dn + n];
}

/* ---------------- fused QKV GEMM (bf16 WMMA, f32 acc) ----------------
 * grid (64, 8, 3), block 256 (8 waves). Block tile 128(M) x 128(N).
 * A/B tiles double-buffered in LDS via async-to-LDS; wave = 32x64.   */
__global__ void __launch_bounds__(256, 1)
gemm_qkv_kernel(const bf16* __restrict__ xh, const bf16* __restrict__ wt,
                const float* __restrict__ bq, const float* __restrict__ bk,
                const float* __restrict__ bv,
                bf16* __restrict__ qh, bf16* __restrict__ kh,
                bf16* __restrict__ vt, float* __restrict__ out) {
  __shared__ bf16 abuf[2][128 * 32];   /* 8 KiB each */
  __shared__ bf16 bbuf[2][128 * 32];

  const int tid  = threadIdx.x;
  const int lane = tid & 31;
  const int w    = tid >> 5;
  const int mat  = blockIdx.z;
  const int row0 = blockIdx.x * 128;
  const int col0 = blockIdx.y * 128;
  const int mw   = w & 3;              /* 4 waves across M */
  const int nw   = w >> 2;             /* 2 waves across N */
  const int ln   = lane & 15;
  const int kb   = (lane >> 4) * 8;

  const bf16*  wtm  = wt + (size_t)mat * Dn * Dn;
  const float* bias = (mat == 0) ? bq : ((mat == 1) ? bk : bv);

  const bf16* asrc = xh  + (size_t)row0 * Dn;   /* row stride Dn */
  const bf16* bsrc = wtm + (size_t)col0 * Dn;

  v8f acc[2][4];
  for (int t = 0; t < 4; ++t) {
    float bval = bias[col0 + nw * 64 + t * 16 + ln];
    for (int r = 0; r < 8; ++r) { acc[0][t][r] = bval; acc[1][t][r] = bval; }
  }

  /* cooperative stage of one 128x32 A tile + 128x32 B tile */
  auto stage = [&](int kk, int buf) {
    for (int i = 0; i < 2; ++i) {
      int c = tid + i * 256;                 /* 512 16B chunks per tile */
      int row = c >> 2, off = (c & 3) * 8;
      async_b128(asrc + (size_t)row * Dn + kk + off, &abuf[buf][row * 32 + off]);
      async_b128(bsrc + (size_t)row * Dn + kk + off, &bbuf[buf][row * 32 + off]);
    }
  };

  stage(0, 0);
  int cur = 0;
  for (int kk = 0; kk < Dn; kk += 32) {
    wait_async();
    __syncthreads();
    if (kk + 32 < Dn) stage(kk + 32, cur ^ 1);

    v16bf a0 = load_frag(&abuf[cur][(mw * 32 + ln) * 32 + kb]);
    v16bf a1 = load_frag(&abuf[cur][(mw * 32 + 16 + ln) * 32 + kb]);
    for (int t = 0; t < 4; ++t) {
      v16bf b = load_frag(&bbuf[cur][(nw * 64 + t * 16 + ln) * 32 + kb]);
      acc[0][t] = wmma_bf16(a0, b, acc[0][t]);
      acc[1][t] = wmma_bf16(a1, b, acc[1][t]);
    }
    cur ^= 1;
  }

  for (int ms = 0; ms < 2; ++ms) {
    const int mbase = row0 + mw * 32 + ms * 16 + 8 * (lane >> 4);
    for (int t = 0; t < 4; ++t) {
      int n = col0 + nw * 64 + t * 16 + ln;
      int h = n >> 6, d = n & 63;
      for (int r = 0; r < 8; ++r) {
        int m = mbase + r;
        int b = m >> 11, s = m & 2047;
        size_t idx = (((size_t)(b * Hn + h) * Sn) + s) * DHn + d;
        float val = acc[ms][t][r];
        if (mat == 0) {
          qh[idx] = (bf16)(val * 0.125f);            /* fold 1/sqrt(64) */
        } else if (mat == 1) {
          out[(size_t)Mn * Dn + idx] = val;          /* K cache f32 */
          kh[idx] = (bf16)val;
        } else {
          out[(size_t)2 * Mn * Dn + idx] = val;      /* V cache f32 */
          vt[((size_t)(b * Hn + h) * DHn + d) * Sn + s] = (bf16)val;
        }
      }
    }
  }
}

/* ---------------- causal flash attention ----------------
 * grid (B*H=64, S/128=16), block 256 (8 waves). Wave = 16 query rows.
 * K/V tiles for each 32-KV block staged once per workgroup into
 * double-buffered LDS (async-to-LDS), shared by all 8 waves.        */
__global__ void __launch_bounds__(256, 1)
attn_kernel(const bf16* __restrict__ qh, const bf16* __restrict__ kh,
            const bf16* __restrict__ vt, float* __restrict__ attn) {
  __shared__ bf16 kbuf[2][32 * 64];    /* 4 KiB each: [kv][dh]  */
  __shared__ bf16 vbuf[2][64 * 32];    /* 4 KiB each: [dh][kv]  */
  __shared__ bf16 pbuf_all[8][16 * 32];

  const int tid  = threadIdx.x;
  const int lane = tid & 31;
  const int w    = tid >> 5;
  const int bh   = blockIdx.x;                 /* b*16 + h */
  const int qblk = blockIdx.y;
  const int q0   = qblk * 128 + w * 16;
  const int ln   = lane & 15;
  const int kb   = (lane >> 4) * 8;

  const bf16* qp = qh + (size_t)bh * Sn * DHn;
  const bf16* kp = kh + (size_t)bh * Sn * DHn;
  const bf16* vp = vt + (size_t)bh * DHn * Sn;

  v16bf qa0 = load_frag(qp + (size_t)(q0 + ln) * DHn + 0  + kb);
  v16bf qa1 = load_frag(qp + (size_t)(q0 + ln) * DHn + 32 + kb);

  v8f acco[4];
  for (int t = 0; t < 4; ++t)
    for (int r = 0; r < 8; ++r) acco[t][r] = 0.0f;
  float mrow[8], lrow[8];
  for (int r = 0; r < 8; ++r) { mrow[r] = -1e30f; lrow[r] = 0.0f; }

  bf16* pbuf = pbuf_all[w];
  const int nblk_wave = (q0 + 47) >> 5;        /* causal bound per wave  */
  const int nblk_blk  = qblk * 4 + 4;          /* uniform over the block */

  /* cooperative stage of one K(32x64) + V(64x32) tile */
  auto stage_kv = [&](int blk, int buf) {
    int kv0 = blk * 32;
    { int row = tid >> 3, off = (tid & 7) * 8;  /* 256 chunks of K */
      async_b128(kp + (size_t)(kv0 + row) * DHn + off,
                 &kbuf[buf][row * 64 + off]); }
    { int row = tid >> 2, off = (tid & 3) * 8;  /* 256 chunks of V^T */
      async_b128(vp + (size_t)row * Sn + kv0 + off,
                 &vbuf[buf][row * 32 + off]); }
  };

  stage_kv(0, 0);
  int cur = 0;
  for (int blk = 0; blk < nblk_blk; ++blk) {
    wait_async();
    __syncthreads();
    if (blk + 1 < nblk_blk) stage_kv(blk + 1, cur ^ 1);

    if (blk < nblk_wave) {
      int kv0 = blk * 32;
      v8f s0, s1;
      for (int r = 0; r < 8; ++r) { s0[r] = 0.0f; s1[r] = 0.0f; }

      v16bf k0a = load_frag(&kbuf[cur][(ln)      * 64 + 0  + kb]);
      v16bf k0b = load_frag(&kbuf[cur][(ln)      * 64 + 32 + kb]);
      s0 = wmma_bf16(qa0, k0a, s0);
      s0 = wmma_bf16(qa1, k0b, s0);
      v16bf k1a = load_frag(&kbuf[cur][(16 + ln) * 64 + 0  + kb]);
      v16bf k1b = load_frag(&kbuf[cur][(16 + ln) * 64 + 32 + kb]);
      s1 = wmma_bf16(qa0, k1a, s1);
      s1 = wmma_bf16(qa1, k1b, s1);

      /* online softmax, per accumulator row r (row = r + 8*(lane>>4)) */
      for (int r = 0; r < 8; ++r) {
        int rowg = q0 + r + 8 * (lane >> 4);
        float v0 = ((kv0 + ln)      <= rowg) ? s0[r] : -1e30f;
        float v1 = ((kv0 + 16 + ln) <= rowg) ? s1[r] : -1e30f;
        float mx = fmaxf(v0, v1);
        for (int off = 1; off < 16; off <<= 1) mx = fmaxf(mx, __shfl_xor(mx, off, 32));
        float mnew = fmaxf(mrow[r], mx);
        float corr = __expf(mrow[r] - mnew);
        float p0 = __expf(v0 - mnew);
        float p1 = __expf(v1 - mnew);
        float rs = p0 + p1;
        for (int off = 1; off < 16; off <<= 1) rs += __shfl_xor(rs, off, 32);
        lrow[r] = lrow[r] * corr + rs;
        mrow[r] = mnew;
        for (int t = 0; t < 4; ++t) acco[t][r] *= corr;
        int mloc = r + 8 * (lane >> 4);
        pbuf[mloc * 32 + ln]      = (bf16)p0;
        pbuf[mloc * 32 + 16 + ln] = (bf16)p1;
      }

      /* P restaged into A-fragment layout (wave-private LDS region) */
      v16bf pa = load_frag(pbuf + ln * 32 + kb);
      for (int t = 0; t < 4; ++t) {
        v16bf bv_ = load_frag(&vbuf[cur][(t * 16 + ln) * 32 + kb]);
        acco[t] = wmma_bf16(pa, bv_, acco[t]);
      }
    }
    cur ^= 1;
  }

  const int b = bh >> 4, h = bh & 15;
  for (int t = 0; t < 4; ++t) {
    for (int r = 0; r < 8; ++r) {
      int mloc = r + 8 * (lane >> 4);
      int s = q0 + mloc;
      attn[((size_t)(b * Sn + s)) * Dn + h * DHn + t * 16 + ln] =
          acco[t][r] / lrow[r];
    }
  }
}

/* ---------------- residual + LayerNorm ---------------- */
__global__ void __launch_bounds__(256, 1)
ln_kernel(const float* __restrict__ x, const float* __restrict__ attn,
          const float* __restrict__ gamma, const float* __restrict__ beta,
          float* __restrict__ out) {
  __shared__ float red[256];
  const int row = blockIdx.x;
  const int tid = threadIdx.x;
  const float* xr = x    + (size_t)row * Dn;
  const float* ar = attn + (size_t)row * Dn;

  float y[4];
  float sum = 0.0f;
  for (int i = 0; i < 4; ++i) {
    y[i] = xr[tid + i * 256] + ar[tid + i * 256];
    sum += y[i];
  }
  red[tid] = sum; __syncthreads();
  for (int s = 128; s > 0; s >>= 1) {
    if (tid < s) red[tid] += red[tid + s];
    __syncthreads();
  }
  float mu = red[0] * (1.0f / 1024.0f);
  __syncthreads();

  float vs = 0.0f;
  for (int i = 0; i < 4; ++i) { float d = y[i] - mu; vs += d * d; }
  red[tid] = vs; __syncthreads();
  for (int s = 128; s > 0; s >>= 1) {
    if (tid < s) red[tid] += red[tid + s];
    __syncthreads();
  }
  float rstd = rsqrtf(red[0] * (1.0f / 1024.0f) + 1e-5f);

  for (int i = 0; i < 4; ++i) {
    int c = tid + i * 256;
    out[(size_t)row * Dn + c] = (y[i] - mu) * rstd * gamma[c] + beta[c];
  }
}

extern "C" void kernel_launch(void* const* d_in, const int* in_sizes, int n_in,
                              void* d_out, int out_size, void* d_ws, size_t ws_size,
                              hipStream_t stream) {
  const float* x     = (const float*)d_in[0];
  const float* Wq    = (const float*)d_in[1];
  const float* bq    = (const float*)d_in[2];
  const float* Wk    = (const float*)d_in[3];
  const float* bk    = (const float*)d_in[4];
  const float* Wv    = (const float*)d_in[5];
  const float* bv    = (const float*)d_in[6];
  const float* gamma = (const float*)d_in[7];
  const float* beta  = (const float*)d_in[8];
  float* out = (float*)d_out;

  char* ws = (char*)d_ws;
  bf16*  xh   = (bf16*)(ws + XH_OFF);
  bf16*  wt   = (bf16*)(ws + WT_OFF);
  bf16*  qh   = (bf16*)(ws + QH_OFF);
  bf16*  kh   = (bf16*)(ws + KH_OFF);
  bf16*  vt   = (bf16*)(ws + VT_OFF);
  float* attn = (float*)(ws + AT_OFF);

  cvt_x_kernel<<<Mn * Dn / 256, 256, 0, stream>>>(x, xh);
  cvt_wT_kernel<<<dim3(Dn * Dn / 256, 3), 256, 0, stream>>>(Wq, Wk, Wv, wt);
  gemm_qkv_kernel<<<dim3(Mn / 128, Dn / 128, 3), 256, 0, stream>>>(
      xh, wt, bq, bk, bv, qh, kh, vt, out);
  attn_kernel<<<dim3(Bn * Hn, Sn / 128), 256, 0, stream>>>(qh, kh, vt, attn);
  ln_kernel<<<Mn, 256, 0, stream>>>(x, attn, gamma, beta, out);
}
